// PromptDetectionLoss_11716670783840
// MI455X (gfx1250) — compile-verified, matrix operand order checked by reading, not run on hardware
//
#include <hip/hip_runtime.h>
#include <hip/hip_bf16.h>

// Availability of the gfx1250 async gather-to-LDS path (probe-confirmed on
// this toolchain: clang-23, generic int* parameters).
#if defined(__has_builtin)
#if __has_builtin(__builtin_amdgcn_global_load_async_to_lds_b32)
#define CDNA5_HAS_ASYNC_LDS 1
#endif
#endif
#ifndef CDNA5_HAS_ASYNC_LDS
#define CDNA5_HAS_ASYNC_LDS 0
#endif

// ---------------------------------------------------------------------------
// Problem constants (match reference setup_inputs)
// ---------------------------------------------------------------------------
namespace {
constexpr int   kB    = 16;     // batch
constexpr int   kN    = 8400;   // anchors
constexpr int   kC    = 80;     // classes
constexpr int   kM    = 24;     // GTs per image
constexpr int   kTopK = 13;
constexpr int   kRM   = 16;     // DFL bins
constexpr float kEps  = 1e-7f;
constexpr float kPi2Inv4 = 0.40528473456935109f;  // 4/pi^2
constexpr int   kNPT  = (kN + 255) / 256;         // anchors per thread (33)
constexpr int   kNPad = kNPT * 256;               // 8448
}

typedef __attribute__((ext_vector_type(16))) _Float16 v16h;
typedef __attribute__((ext_vector_type(8)))  float    v8f;

__device__ __forceinline__ float sigmf(float x) { return 1.0f / (1.0f + expf(-x)); }
__device__ __forceinline__ int   iminf(int a, int b) { return a < b ? a : b; }
__device__ __forceinline__ int   imaxf(int a, int b) { return a > b ? a : b; }

__device__ __forceinline__ void wait_async_zero() {
#if defined(__has_builtin) && __has_builtin(__builtin_amdgcn_s_wait_asynccnt)
  __builtin_amdgcn_s_wait_asynccnt(0);
#else
  asm volatile("s_wait_asynccnt 0" ::: "memory");
#endif
  asm volatile("" ::: "memory");  // keep LDS reads after the wait
}

// Pairwise IoU between boxes a=(a0..a3) and b=(b0..b3), reference-faithful.
__device__ __forceinline__ float iou_pair(float a0, float a1, float a2, float a3,
                                          float b0, float b1, float b2, float b3) {
  float iw = fmaxf(fminf(a2, b2) - fmaxf(a0, b0), 0.0f);
  float ih = fmaxf(fminf(a3, b3) - fmaxf(a1, b1), 0.0f);
  float inter = iw * ih;
  float areaA = fmaxf(a2 - a0, 0.0f) * fmaxf(a3 - a1, 0.0f);
  float areaB = fmaxf(b2 - b0, 0.0f) * fmaxf(b3 - b1, 0.0f);
  return inter / (areaA + areaB - inter + kEps);
}

// ---------------------------------------------------------------------------
// Kernel A: per-(image, GT) align metric + top-13 selection.
// One 256-thread block per (b,g). Score column gather goes through the CDNA5
// async gather-to-LDS path (ASYNCcnt) when available; align cached in LDS;
// 13 block-argmax rounds.
// ---------------------------------------------------------------------------
__global__ void __launch_bounds__(256)
assign_topk_kernel(const float* __restrict__ pred_boxes,
                   const float* __restrict__ pred_scores,
                   const float* __restrict__ anchor_points,
                   const unsigned char* __restrict__ class_mask,
                   const float* __restrict__ gt_boxes,
                   const int*   __restrict__ gt_labels,
                   float* __restrict__ topv, int* __restrict__ topi) {
  __shared__ float alignBuf[kNPad];
  __shared__ float rv[256];
  __shared__ int   ri[256];

  const int b = blockIdx.x / kM;
  const int g = blockIdx.x % kM;
  const int tid = threadIdx.x;
  const size_t gi = (size_t)b * kM + g;
  const float4* __restrict__ pbox4 = reinterpret_cast<const float4*>(pred_boxes);
  const float4  gbox = reinterpret_cast<const float4*>(gt_boxes)[gi];
  const float b0 = gbox.x, b1 = gbox.y, b2 = gbox.z, b3 = gbox.w;

  const int lab  = gt_labels[gi];
  const int labc = iminf(imaxf(lab, 0), kC - 1);
  const bool valid = (lab >= 0) && (lab < kC) && (class_mask[b * kC + labc] != 0);

#if CDNA5_HAS_ASYNC_LDS
  // Pass 1: IoU^6 in registers; kick off per-lane async gathers of the score
  // column into alignBuf (hardware gather-to-LDS, tracked by ASYNCcnt).
  float pre[kNPT];
#pragma unroll
  for (int i = 0; i < kNPT; ++i) {
    const int n = tid + i * 256;
    float val = -2.0f;
    if (n < kN) {
      val = -1.0f;
      if (valid) {
        const float ax = anchor_points[n * 2 + 0];
        const float ay = anchor_points[n * 2 + 1];
        if (ax >= b0 && ax <= b2 && ay >= b1 && ay <= b3) {
          const float4 p = pbox4[(size_t)b * kN + n];
          const float iou = iou_pair(p.x, p.y, p.z, p.w, b0, b1, b2, b3);
          const float i2 = iou * iou;
          val = i2 * i2 * i2;  // >= 0 marks "score pending"
          const float* gsrc = pred_scores + ((size_t)b * kN + n) * kC + labc;
          // clang-23 declares this builtin with generic int* parameters.
          __builtin_amdgcn_global_load_async_to_lds_b32(
              (int*)gsrc, (int*)&alignBuf[n], 0, 0);
        }
      }
    }
    pre[i] = val;
  }
  wait_async_zero();
  // Pass 2: combine staged raw score with IoU^6 (each thread owns slot n).
#pragma unroll
  for (int i = 0; i < kNPT; ++i) {
    const int n = tid + i * 256;
    float val = pre[i];
    if (val >= 0.0f) val *= sigmf(alignBuf[n]);
    alignBuf[n] = val;
  }
#else
  for (int i = 0; i < kNPT; ++i) {
    const int n = tid + i * 256;
    float val = -2.0f;
    if (n < kN) {
      val = -1.0f;
      if (valid) {
        const float ax = anchor_points[n * 2 + 0];
        const float ay = anchor_points[n * 2 + 1];
        if (ax >= b0 && ax <= b2 && ay >= b1 && ay <= b3) {
          const float4 p = pbox4[(size_t)b * kN + n];
          const float iou = iou_pair(p.x, p.y, p.z, p.w, b0, b1, b2, b3);
          const float s = pred_scores[((size_t)b * kN + n) * kC + labc];
          const float i2 = iou * iou;
          val = sigmf(s) * (i2 * i2 * i2);  // cls^1 * iou^6
        }
      }
    }
    alignBuf[n] = val;
  }
#endif
  __syncthreads();

  // 13 rounds of block-wide argmax (ties -> lowest anchor index, like top_k).
  for (int k = 0; k < kTopK; ++k) {
    float bv = -3.0e38f;
    int   bi = 0x7fffffff;
    for (int i = 0; i < kNPT; ++i) {
      const int n = tid + i * 256;
      const float v = alignBuf[n];
      if (v > bv) { bv = v; bi = n; }
    }
    rv[tid] = bv; ri[tid] = bi;
    __syncthreads();
    for (int s = 128; s > 0; s >>= 1) {
      if (tid < s) {
        const float v2 = rv[tid + s];
        const int   i2 = ri[tid + s];
        if (v2 > rv[tid] || (v2 == rv[tid] && i2 < ri[tid])) { rv[tid] = v2; ri[tid] = i2; }
      }
      __syncthreads();
    }
    if (tid == 0) {
      topv[gi * kTopK + k] = rv[0];
      topi[gi * kTopK + k] = ri[0];
      alignBuf[ri[0]] = -2.0f;  // mark used
    }
    __syncthreads();
  }
}

// ---------------------------------------------------------------------------
// Kernel B: claim resolution. key = (bits(align) << 32) | (M - g).
// Positive-float bit patterns are order-preserving; low word gives the
// reference's strict-> (earliest-GT-wins-ties) semantics under atomicMax.
// ---------------------------------------------------------------------------
__global__ void resolve_claims_kernel(const float* __restrict__ topv,
                                      const int*   __restrict__ topi,
                                      unsigned long long* __restrict__ claim) {
  const int e = blockIdx.x * blockDim.x + threadIdx.x;
  if (e >= kB * kM * kTopK) return;
  const float v = topv[e];
  if (v < 0.0f) return;  // masked / invalid candidates never claim
  const int b = e / (kM * kTopK);
  const int g = (e / kTopK) % kM;
  const int n = topi[e];
  const unsigned long long key =
      ((unsigned long long)__float_as_uint(v) << 32) |
      (unsigned long long)(unsigned)(kM - g);
  atomicMax(&claim[(size_t)b * kN + n], key);
}

// ---------------------------------------------------------------------------
// Matrix-pipe cross-wave reduction: 8 waves x 8 channels summed via one
// v_wmma_f32_16x16x32_f16 with an exact hi/lo f16 split (f32-accurate).
// Channel layout in A: row m = channel, K = wave for hi, K+8 = wave for lo.
// B = all-ones. Lane 0's D[0..7][0] are the channel sums (f32 accumulate).
// ---------------------------------------------------------------------------
__device__ __forceinline__ void block_reduce_wmma(float ch[8], float* wpart,
                                                  const int tid,
                                                  float* __restrict__ accB) {
#pragma unroll
  for (int off = 16; off > 0; off >>= 1) {
#pragma unroll
    for (int c = 0; c < 8; ++c) ch[c] += __shfl_xor(ch[c], off, 32);
  }
  const int lane = tid & 31, wid = tid >> 5;
  if (lane == 0) {
#pragma unroll
    for (int c = 0; c < 8; ++c) wpart[wid * 8 + c] = ch[c];
  }
  __syncthreads();
  if (tid < 32) {  // wave 0, EXEC all-ones (block is always a full 256 threads)
    const int m = tid & 15;
    v16h a, ones;
#pragma unroll
    for (int i = 0; i < 16; ++i) { a[i] = (_Float16)0.0f; ones[i] = (_Float16)1.0f; }
#pragma unroll
    for (int w = 0; w < 8; ++w) {
      const float v = (m < 8) ? wpart[w * 8 + m] : 0.0f;
      const _Float16 hi = (_Float16)v;
      const float rem = v - (float)hi;
      // lanes 0-15 hold K=0..7 at a[0..7] (hi), lanes 16-31 hold K=8..15 (lo)
      a[w] = (tid < 16) ? hi : (_Float16)rem;
    }
    v8f d = {};
    d = __builtin_amdgcn_wmma_f32_16x16x32_f16(false, a, false, ones,
                                               (short)0, d, false, false);
    if (tid == 0) {
#pragma unroll
      for (int c = 0; c < 8; ++c) atomicAdd(&accB[c], d[c]);
    }
  }
  __syncthreads();
}

// ---------------------------------------------------------------------------
// Kernel C: fused per-anchor losses + statistics.
// Channels: 0 obj-focal, 1 (1-ciou)*fg, 2 CE*fg, 3 dfl*fg,
//           4 fg count, 5 iou*fg, 6 pos_joint*fg, 7 obj*neg_best*(1-fg)
// ---------------------------------------------------------------------------
__global__ void __launch_bounds__(256)
loss_kernel(const float* __restrict__ pred_boxes,
            const float* __restrict__ pred_scores,
            const float* __restrict__ pred_obj,
            const float* __restrict__ anchor_points,
            const float* __restrict__ stride_tensor,
            const float* __restrict__ box_dist,
            const unsigned char* __restrict__ class_mask,
            const float* __restrict__ gt_boxes,
            const int*   __restrict__ gt_labels,
            const unsigned long long* __restrict__ claim,
            float* __restrict__ acc) {
  __shared__ unsigned char maskRow[kC];
  __shared__ float wpart[8 * 8];

  const int b = blockIdx.y;
  const int tid = threadIdx.x;
  const int n = blockIdx.x * 256 + tid;
  if (tid < kC) maskRow[tid] = class_mask[b * kC + tid];
  __syncthreads();

  float ch[8];
#pragma unroll
  for (int c = 0; c < 8; ++c) ch[c] = 0.0f;

  if (n < kN) {
    const size_t base = (size_t)b * kN + n;
    const unsigned long long key = claim[base];
    const bool fg = (key != 0ull);
    const int g = fg ? (kM - (int)(key & 0xffffffffull)) : 0;

    const float4 pbx = reinterpret_cast<const float4*>(pred_boxes)[base];
    const float a0 = pbx.x, a1 = pbx.y, a2 = pbx.z, a3 = pbx.w;
    float t0 = a0, t1 = a1, t2 = a2, t3 = a3;
    int lab = 0;
    if (fg) {
      const size_t gi = (size_t)b * kM + g;
      const float4 tbx = reinterpret_cast<const float4*>(gt_boxes)[gi];
      t0 = tbx.x; t1 = tbx.y; t2 = tbx.z; t3 = tbx.w;
      lab = iminf(imaxf(gt_labels[gi], 0), kC - 1);
    }

    // --- objectness focal loss ---
    const float x  = pred_obj[base];
    const float tf = fg ? 1.0f : 0.0f;
    const float objsig = sigmf(x);
    const float ce_obj = fmaxf(x, 0.0f) - x * tf + log1pf(expf(-fabsf(x)));
    const float p_t = objsig * tf + (1.0f - objsig) * (1.0f - tf);
    const float af  = 0.25f * tf + 0.75f * (1.0f - tf);
    const float om  = 1.0f - p_t;
    ch[0] = ce_obj * af * om * om;

    // --- IoU / CIoU (fg only contributes) ---
    const float iou = iou_pair(a0, a1, a2, a3, t0, t1, t2, t3);
    if (fg) {
      const float cw = fmaxf(a2, t2) - fminf(a0, t0);
      const float chh = fmaxf(a3, t3) - fminf(a1, t1);
      const float c2 = cw * cw + chh * chh + kEps;
      const float dx = a0 + a2 - t0 - t2, dy = a1 + a3 - t1 - t3;
      const float rho2 = (dx * dx + dy * dy) * 0.25f;
      const float w1 = a2 - a0, h1 = a3 - a1, w2 = t2 - t0, h2 = t3 - t1;
      const float dv = atanf(w2 / (h2 + kEps)) - atanf(w1 / (h1 + kEps));
      const float v = kPi2Inv4 * dv * dv;
      const float alpha = v / (v - iou + 1.0f + kEps);
      const float ciou = iou - rho2 / c2 - v * alpha;
      ch[1] = 1.0f - ciou;
      ch[4] = 1.0f;
      ch[5] = iou;
    }

    // --- masked log-softmax over classes (online logsumexp) ---
    const float* row = pred_scores + base * kC;
    __builtin_prefetch(row, 0, 1);
    float mx = -3.0e38f, se = 0.0f, slab = 0.0f;
    for (int c = 0; c < kC; ++c) {
      const float s = row[c];
      const float msv = maskRow[c] ? s : -10000.0f;
      if (fg && c == lab) slab = msv;
      if (msv > mx) { se = se * expf(mx - msv) + 1.0f; mx = msv; }
      else          { se += expf(msv - mx); }
    }
    const float lse = logf(se) + mx;
    if (fg) {
      ch[2] = lse - slab;                  // CE on matched class
      ch[6] = objsig * sigmf(slab);        // pos_joint
    } else {
      ch[7] = objsig * sigmf(mx);          // obj * neg_best (sigmoid monotone)
    }

    // --- DFL (fg only; skips 64-float row for ~97% of anchors) ---
    if (fg) {
      const float ax = anchor_points[n * 2 + 0];
      const float ay = anchor_points[n * 2 + 1];
      const float sv = stride_tensor[n];
      const float dd[4] = { (ax - t0) / sv, (ay - t1) / sv,
                            (t2 - ax) / sv, (t3 - ay) / sv };
      const float* dr = box_dist + base * (4 * kRM);
      __builtin_prefetch(dr, 0, 1);
      float dfl = 0.0f;
#pragma unroll
      for (int j = 0; j < 4; ++j) {
        const float tg = fminf(fmaxf(dd[j], 0.0f), (float)(kRM - 1) - 0.01f);
        const int tl = (int)tg;
        const int tr = iminf(tl + 1, kRM - 1);
        const float wl = (float)tr - tg, wr = 1.0f - wl;
        float m2 = -3.0e38f;
        for (int k = 0; k < kRM; ++k) m2 = fmaxf(m2, dr[j * kRM + k]);
        float s2 = 0.0f, vl = 0.0f, vr = 0.0f;
        for (int k = 0; k < kRM; ++k) {
          const float v = dr[j * kRM + k];
          s2 += expf(v - m2);
          if (k == tl) vl = v;
          if (k == tr) vr = v;
        }
        const float l2 = logf(s2) + m2;
        dfl += (l2 - vl) * wl + (l2 - vr) * wr;
      }
      ch[3] = dfl;
    }
  }

  block_reduce_wmma(ch, wpart, tid, acc + b * 8);
}

// ---------------------------------------------------------------------------
// Kernel D: finalize 10 output scalars from per-image accumulators.
// ---------------------------------------------------------------------------
__global__ void finalize_kernel(const float* __restrict__ acc, float* __restrict__ out) {
  if (threadIdx.x != 0 || blockIdx.x != 0) return;
  float t_obj = 0.f, t_iou = 0.f, t_match = 0.f, t_dfl = 0.f;
  float t_pos = 0.f, pos_s = 0.f, neg_s = 0.f, miou_s = 0.f;
  for (int b = 0; b < kB; ++b) {
    const float* a = acc + b * 8;
    const float cnt = a[4];
    const float cs = fmaxf(cnt, 1.0f);
    t_obj   += a[0] / (float)kN;      // mean over points per image
    t_iou   += a[1] / cs;
    t_match += a[2] / cs;
    t_dfl   += a[3] / (4.0f * cs);
    t_pos   += cnt;
    miou_s  += a[5];
    pos_s   += a[6];
    neg_s   += a[7];
  }
  const float t_neg = (float)kB * (float)kN - t_pos;
  const float nb = (float)kB;
  out[0] = (1.0f * t_obj + 1.0f * t_match + 7.5f * t_iou + 1.5f * t_dfl) / nb;
  out[1] = t_obj / nb;
  out[2] = t_match / nb;
  out[3] = t_iou / nb;
  out[4] = t_dfl / nb;
  out[5] = t_pos;
  out[6] = t_neg;
  out[7] = pos_s / fmaxf(t_pos, 1.0f);
  out[8] = neg_s / fmaxf(t_neg, 1.0f);
  out[9] = miou_s / fmaxf(t_pos, 1.0f);
}

// ---------------------------------------------------------------------------
// Host entry
// ---------------------------------------------------------------------------
extern "C" void kernel_launch(void* const* d_in, const int* in_sizes, int n_in,
                              void* d_out, int out_size, void* d_ws, size_t ws_size,
                              hipStream_t stream) {
  (void)in_sizes; (void)n_in; (void)out_size; (void)ws_size;
  const float* pred_boxes    = (const float*)d_in[0];   // [B,N,4]
  const float* pred_scores   = (const float*)d_in[1];   // [B,N,C]
  const float* pred_obj      = (const float*)d_in[2];   // [B,N]
  const float* anchor_points = (const float*)d_in[3];   // [N,2]
  const float* stride_tensor = (const float*)d_in[4];   // [N,1]
  const float* box_dist      = (const float*)d_in[5];   // [B,N,64]
  const unsigned char* class_mask = (const unsigned char*)d_in[6];  // [B,C] bool
  const float* gt_boxes      = (const float*)d_in[7];   // [B,M,4]
  const int*   gt_labels     = (const int*)d_in[8];     // [B,M]
  float* out = (float*)d_out;

  // Workspace layout
  char* ws = (char*)d_ws;
  unsigned long long* claim = (unsigned long long*)ws;
  size_t off = (size_t)kB * kN * sizeof(unsigned long long);
  float* topv = (float*)(ws + off); off += (size_t)kB * kM * kTopK * sizeof(float);
  int*   topi = (int*)(ws + off);   off += (size_t)kB * kM * kTopK * sizeof(int);
  float* acc  = (float*)(ws + off); off += (size_t)kB * 8 * sizeof(float);

  (void)hipMemsetAsync(d_ws, 0, off, stream);

  assign_topk_kernel<<<kB * kM, 256, 0, stream>>>(
      pred_boxes, pred_scores, anchor_points, class_mask, gt_boxes, gt_labels,
      topv, topi);

  const int tot = kB * kM * kTopK;
  resolve_claims_kernel<<<(tot + 255) / 256, 256, 0, stream>>>(topv, topi, claim);

  dim3 gridC((kN + 255) / 256, kB);
  loss_kernel<<<gridC, 256, 0, stream>>>(
      pred_boxes, pred_scores, pred_obj, anchor_points, stride_tensor, box_dist,
      class_mask, gt_boxes, gt_labels, claim, acc);

  finalize_kernel<<<1, 32, 0, stream>>>(acc, out);
}
